// NextVLAD_16449724743929
// MI455X (gfx1250) — matrix operand order repaired and checked by reading, not compiled, not run on hardware
//
#include <hip/hip_runtime.h>

typedef _Float16 f16;
typedef __attribute__((ext_vector_type(8)))  _Float16 v8h;
typedef __attribute__((ext_vector_type(16))) _Float16 v16h;
typedef __attribute__((ext_vector_type(8)))  float    v8f;

#define TILE_M 128
#define TILE_N 128
#define TILE_K 32
#define LDSTRIDE 40          // 32 + 8 halfs pad, keeps 16B alignment (80B rows)
#define MI 4                 // 16-row tiles per wave (M)
#define NJ 2                 // 16-col tiles per wave (N)

#define FLAG_RELU 1
#define FLAG_SIG  2

__device__ __forceinline__ float waveSum(float v) {
#pragma unroll
  for (int m = 16; m > 0; m >>= 1) v += __shfl_xor(v, m, 32);
  return v;
}
__device__ __forceinline__ float waveMax(float v) {
#pragma unroll
  for (int m = 16; m > 0; m >>= 1) v = fmaxf(v, __shfl_xor(v, m, 32));
  return v;
}

// ---------------------------------------------------------------------------
// Generic NT GEMM: C[M,N] = A[M,K] @ Bt[N,K]^T  (f16 in, f32 accum via WMMA)
// 128x128 WG tile, 8 waves (2x4), 64x32 per wave = 8 WMMA/iter.
// Double-buffered LDS, one barrier per K step, branch-free hot loop
// (row clamp instead of zero-fill: clamped rows only feed unstored C rows).
// Epilogue: +bias, BatchNorm, -subA*subB (VLAD residual), relu/sigmoid,
// dual f32/f16 stores. Optional batch strides (gridDim.z).
// ---------------------------------------------------------------------------
__global__ __launch_bounds__(256) void gemm_nt_wmma(
    const f16* __restrict__ A, const f16* __restrict__ Bt,
    const float* __restrict__ bias,
    const float* __restrict__ bn_g, const float* __restrict__ bn_b,
    const float* __restrict__ bn_m, const float* __restrict__ bn_v,
    const float* __restrict__ subA, const float* __restrict__ subB,
    float* __restrict__ outF, f16* __restrict__ outH,
    int M, int N, int Kd,
    long sAb, long sBb, long sCb, int flags)
{
  __shared__ f16 As[2][TILE_M][LDSTRIDE];
  __shared__ f16 Bs[2][TILE_N][LDSTRIDE];

  const int bz = blockIdx.z;
  A  += (long)bz * sAb;
  Bt += (long)bz * sBb;
  const long cbase = (long)bz * sCb;

  const int tid  = threadIdx.x;
  const int wave = tid >> 5;
  const int lane = tid & 31;
  const int wrow = wave >> 2;         // 2 wave-rows  -> 64 M rows each
  const int wcol = wave & 3;          // 4 wave-cols  -> 32 N cols each
  const int half = lane >> 4;         // hi/lo 16-lane group
  const int l    = lane & 15;

  const int m0 = blockIdx.x * TILE_M + wrow * 64;
  const int n0 = blockIdx.y * TILE_N + wcol * 32;

  // ---- per-thread staging slots (branch-free, row-clamped) ----
  // 128 rows x 4 chunks(8h) = 512 chunks per matrix; 256 threads x 2 chunks.
  int arow[2], acol[2], brow[2], bcol[2];
  const f16* aptr[2];
  const f16* bptr[2];
#pragma unroll
  for (int q = 0; q < 2; ++q) {
    const int c = tid + q * 256;
    const int r = c >> 2, cc = (c & 3) * 8;
    arow[q] = r; acol[q] = cc;
    brow[q] = r; bcol[q] = cc;
    int ga = blockIdx.x * TILE_M + r; ga = (ga < M) ? ga : (M - 1);
    int gb = blockIdx.y * TILE_N + r; gb = (gb < N) ? gb : (N - 1);
    aptr[q] = A + (long)ga * Kd + cc;
    bptr[q] = Bt + (long)gb * Kd + cc;
  }

  v8f acc[MI][NJ] = {};

  const int nkt = Kd / TILE_K;

  // ---- preload K-tile 0 into LDS buffer 0 ----
  {
    v8h ra[2], rb[2];
#pragma unroll
    for (int q = 0; q < 2; ++q) {
      ra[q] = *(const v8h*)(aptr[q]);
      rb[q] = *(const v8h*)(bptr[q]);
    }
#pragma unroll
    for (int q = 0; q < 2; ++q) {
      *(v8h*)(&As[0][arow[q]][acol[q]]) = ra[q];
      *(v8h*)(&Bs[0][brow[q]][bcol[q]]) = rb[q];
    }
  }
  __syncthreads();

  for (int kt = 0; kt < nkt; ++kt) {
    const int cur = kt & 1, nxt = cur ^ 1;
    const int ktn = (kt + 1 < nkt) ? (kt + 1) : (nkt - 1);   // clamp: tail re-reads
    const long kbn = (long)ktn * TILE_K;

    // issue next tile's global loads early (latency hidden by WMMA block)
    v8h ra[2], rb[2];
#pragma unroll
    for (int q = 0; q < 2; ++q) {
      ra[q] = *(const v8h*)(aptr[q] + kbn);
      rb[q] = *(const v8h*)(bptr[q] + kbn);
      __builtin_prefetch(aptr[q] + kbn + TILE_K, 0, 1);
      __builtin_prefetch(bptr[q] + kbn + TILE_K, 0, 1);
    }

    // ---- fragments per CDNA5 16-bit WMMA VGPR layout ----
    v16h afrag[MI], bfrag[NJ];
#pragma unroll
    for (int i = 0; i < MI; ++i) {
      const f16* ap = &As[cur][wrow * 64 + i * 16 + l][half * 8];
      v8h lo = *(const v8h*)(ap);        // K = half*8 .. half*8+7
      v8h hi = *(const v8h*)(ap + 16);   // K = 16+half*8 ..
      v16h t;
#pragma unroll
      for (int e = 0; e < 8; ++e) { t[e] = lo[e]; t[8 + e] = hi[e]; }
      afrag[i] = t;
    }
#pragma unroll
    for (int j = 0; j < NJ; ++j) {
      const f16* bp = &Bs[cur][wcol * 32 + j * 16 + l][half * 8];
      v8h lo = *(const v8h*)(bp);
      v8h hi = *(const v8h*)(bp + 16);
      v16h t;
#pragma unroll
      for (int e = 0; e < 8; ++e) { t[e] = lo[e]; t[8 + e] = hi[e]; }
      bfrag[j] = t;
    }

#pragma unroll
    for (int i = 0; i < MI; ++i)
#pragma unroll
      for (int j = 0; j < NJ; ++j)
        acc[i][j] = __builtin_amdgcn_wmma_f32_16x16x32_f16(
            false, afrag[i], false, bfrag[j], (short)0, acc[i][j], false, false);

    // stage next tile into the other LDS buffer
#pragma unroll
    for (int q = 0; q < 2; ++q) {
      *(v8h*)(&As[nxt][arow[q]][acol[q]]) = ra[q];
      *(v8h*)(&Bs[nxt][brow[q]][bcol[q]]) = rb[q];
    }
    __syncthreads();
  }

  // ---- epilogue: C layout = VGPR r -> M=r (lanes 0-15) / M=r+8 (lanes 16-31)
#pragma unroll
  for (int i = 0; i < MI; ++i) {
#pragma unroll
    for (int j = 0; j < NJ; ++j) {
      const int col = n0 + j * 16 + l;
      if (col >= N) continue;
#pragma unroll
      for (int r = 0; r < 8; ++r) {
        const int row = m0 + i * 16 + half * 8 + r;
        if (row >= M) continue;
        float v = acc[i][j][r];
        if (bias) v += bias[col];
        if (bn_g) v = (v - bn_m[col]) * rsqrtf(bn_v[col] + 1e-5f) * bn_g[col] + bn_b[col];
        if (subA) v -= subA[(long)bz * N + col] * subB[(long)row * N + col];
        if (flags & FLAG_RELU) v = fmaxf(v, 0.0f);
        if (flags & FLAG_SIG)  v = 1.0f / (1.0f + __expf(-v));
        const long cidx = cbase + (long)row * N + col;
        if (outF) outF[cidx] = v;
        if (outH) outH[cidx] = (f16)v;
      }
    }
  }
}

// ---------------------------------------------------------------------------
// Row L2-normalize x [rows, ncols] f32 -> f16
// ---------------------------------------------------------------------------
__global__ __launch_bounds__(256) void l2norm_rows(
    const float* __restrict__ x, f16* __restrict__ out, int ncols)
{
  const long row = blockIdx.x;
  const float* xr = x + row * (long)ncols;
  const int wave = threadIdx.x >> 5, lane = threadIdx.x & 31;
  __shared__ float red[8];

  float s = 0.0f;
  for (int c = threadIdx.x; c < ncols; c += 256) { float v = xr[c]; s += v * v; }
  s = waveSum(s);
  if (lane == 0) red[wave] = s;
  __syncthreads();
  if (wave == 0) {
    float t = (lane < 8) ? red[lane] : 0.0f;
    t = waveSum(t);
    if (lane == 0) red[0] = t;
  }
  __syncthreads();
  const float inv = 1.0f / fmaxf(sqrtf(red[0]), 1e-12f);
  f16* orow = out + row * (long)ncols;
  for (int c = threadIdx.x; c < ncols; c += 256) orow[c] = (f16)(xr[c] * inv);
}

// ---------------------------------------------------------------------------
__global__ void cvt_f16_kernel(const float* __restrict__ in, f16* __restrict__ out, long n)
{
  long i = (long)blockIdx.x * blockDim.x + threadIdx.x;
  const long stride = (long)gridDim.x * blockDim.x;
  for (; i < n; i += stride) out[i] = (f16)in[i];
}

// in [Rn, Cn] f32 -> out [Cn, Rn] f16
__global__ void transpose_cvt_kernel(const float* __restrict__ in, f16* __restrict__ out,
                                     int Rn, int Cn)
{
  long i = (long)blockIdx.x * blockDim.x + threadIdx.x;
  const long n = (long)Rn * Cn;
  const long stride = (long)gridDim.x * blockDim.x;
  for (; i < n; i += stride) {
    const long c = i / Rn, r = i % Rn;
    out[i] = (f16)in[r * (long)Cn + c];
  }
}

// ---------------------------------------------------------------------------
// att[m,g] = sigmoid(sum_d h[m,d]*fc2_w[g,d] + fc2_b[g]); one wave per g
// ---------------------------------------------------------------------------
__global__ __launch_bounds__(256) void att_kernel(
    const f16* __restrict__ h, const float* __restrict__ w,
    const float* __restrict__ b, float* __restrict__ att, int Kd)
{
  const long m = blockIdx.x;
  const int g = threadIdx.x >> 5, lane = threadIdx.x & 31;
  const f16* hr = h + m * (long)Kd;
  const float* wr = w + g * (long)Kd;
  float s = 0.0f;
  for (int d = lane; d < Kd; d += 32) s += (float)hr[d] * wr[d];
  s = waveSum(s);
  if (lane == 0) att[m * 8 + g] = 1.0f / (1.0f + __expf(-(s + b[g])));
}

// ---------------------------------------------------------------------------
// segment softmax (K=64) * attention, in-place on act [9600,512];
// also scatter into actT [B][64][2400]
// ---------------------------------------------------------------------------
__global__ __launch_bounds__(256) void softmax_att_kernel(
    float* __restrict__ act, const float* __restrict__ att, f16* __restrict__ actT)
{
  const long seg = (long)blockIdx.x * 8 + (threadIdx.x >> 5);   // 76800 segments
  const int lane = threadIdx.x & 31;
  const long m = seg >> 3;          // row in [0,9600)
  const int g = (int)(seg & 7);
  float* p = act + m * 512 + g * 64;
  float v0 = p[lane], v1 = p[lane + 32];
  const float mx = waveMax(fmaxf(v0, v1));
  float e0 = __expf(v0 - mx), e1 = __expf(v1 - mx);
  const float sum = waveSum(e0 + e1);
  const float at = att[m * 8 + g] / sum;
  const float r0 = e0 * at, r1 = e1 * at;
  p[lane] = r0;
  p[lane + 32] = r1;
  const long b = m / 300, fr = m - b * 300, n = fr * 8 + g;
  f16* t = actT + b * 64 * 2400 + n;
  t[(long)lane * 2400]        = (f16)r0;
  t[(long)(lane + 32) * 2400] = (f16)r1;
}

// asum[b,k] = sum_n act_soft[b,n,k]; one wave per (b,k)
__global__ __launch_bounds__(256) void colsum_kernel(
    const float* __restrict__ act, float* __restrict__ asum)
{
  const long id = (long)blockIdx.x * 8 + (threadIdx.x >> 5);    // 2048 pairs
  const int lane = threadIdx.x & 31;
  const int b = (int)(id >> 6), k = (int)(id & 63);
  float s = 0.0f;
  for (int n = lane; n < 2400; n += 32) {
    const int fr = n >> 3, g = n & 7;
    s += act[((long)b * 300 + fr) * 512 + g * 64 + k];
  }
  s = waveSum(s);
  if (lane == 0) asum[b * 64 + k] = s;
}

// resT[b][f][n] = h[b*300 + n/8, (n%8)*512 + f]
__global__ void build_resT_kernel(const f16* __restrict__ h, f16* __restrict__ resT)
{
  const long total = 32L * 512 * 2400;
  long i = (long)blockIdx.x * blockDim.x + threadIdx.x;
  const long stride = (long)gridDim.x * blockDim.x;
  for (; i < total; i += stride) {
    const long n = i % 2400, t = i / 2400, f = t % 512, b = t / 512;
    const long fr = n >> 3, g = n & 7;
    resT[i] = h[((long)b * 300 + fr) * 4096 + g * 512 + f];
  }
}

// intra-norm over FEAT (axis f) for each (b,k), then bn2, store f16 feat-major
__global__ __launch_bounds__(256) void intranorm_kernel(
    const float* __restrict__ vlad,
    const float* __restrict__ g2, const float* __restrict__ b2,
    const float* __restrict__ m2, const float* __restrict__ v2,
    f16* __restrict__ out)
{
  const long id = (long)blockIdx.x * 8 + (threadIdx.x >> 5);    // 2048 (b,k) pairs
  const int lane = threadIdx.x & 31;
  const int bb = (int)(id >> 6), k = (int)(id & 63);
  const float* vp = vlad + (long)bb * 512 * 64 + k;
  float s = 0.0f;
  for (int f = lane; f < 512; f += 32) { const float x = vp[(long)f * 64]; s += x * x; }
  s = waveSum(s);
  const float inv = 1.0f / fmaxf(sqrtf(s), 1e-12f);
  for (int f = lane; f < 512; f += 32) {
    const int idx = f * 64 + k;
    float x = vp[(long)f * 64] * inv;
    x = (x - m2[idx]) * rsqrtf(v2[idx] + 1e-5f) * g2[idx] + b2[idx];
    out[(long)bb * 32768 + idx] = (f16)x;
  }
}

__global__ void mulcvt_kernel(const float* __restrict__ y, const float* __restrict__ gate,
                              f16* __restrict__ out, long n)
{
  long i = (long)blockIdx.x * blockDim.x + threadIdx.x;
  const long stride = (long)gridDim.x * blockDim.x;
  for (; i < n; i += stride) out[i] = (f16)(y[i] * gate[i]);
}

// ---------------------------------------------------------------------------
extern "C" void kernel_launch(void* const* d_in, const int* in_sizes, int n_in,
                              void* d_out, int out_size, void* d_ws, size_t ws_size,
                              hipStream_t stream)
{
  // dims
  const int Bb = 32, D = 2048, ED = 4096;
  const int FEAT = 512, VLAD = 32768, H = 2048, HR = 256, NC = 3862;
  const long MR = 9600;               // B*FR

  // inputs (setup_inputs order)
  const float* x      = (const float*)d_in[0];
  const float* fc1_w  = (const float*)d_in[1];
  const float* fc1_b  = (const float*)d_in[2];
  const float* fc2_w  = (const float*)d_in[3];
  const float* fc2_b  = (const float*)d_in[4];
  const float* c1     = (const float*)d_in[5];
  const float* c2     = (const float*)d_in[6];
  const float* bn1_g  = (const float*)d_in[7];
  const float* bn1_b  = (const float*)d_in[8];
  const float* bn1_m  = (const float*)d_in[9];
  const float* bn1_v  = (const float*)d_in[10];
  const float* bn2_g  = (const float*)d_in[11];
  const float* bn2_b  = (const float*)d_in[12];
  const float* bn2_m  = (const float*)d_in[13];
  const float* bn2_v  = (const float*)d_in[14];
  const float* cgf_w  = (const float*)d_in[15];
  const float* cgf_b  = (const float*)d_in[16];
  const float* cgbn_g = (const float*)d_in[17];
  const float* cgbn_b = (const float*)d_in[18];
  const float* cgbn_m = (const float*)d_in[19];
  const float* cgbn_v = (const float*)d_in[20];
  const float* g1_w   = (const float*)d_in[21];
  const float* g1_b   = (const float*)d_in[22];
  const float* gbn_g  = (const float*)d_in[23];
  const float* gbn_b  = (const float*)d_in[24];
  const float* gbn_m  = (const float*)d_in[25];
  const float* gbn_v  = (const float*)d_in[26];
  const float* g2_w   = (const float*)d_in[27];
  const float* g2_b   = (const float*)d_in[28];
  const float* fc3_w  = (const float*)d_in[29];
  const float* fc3_b  = (const float*)d_in[30];

  // workspace layout
  size_t off = 0;
  auto alloc = [&](size_t bytes) -> void* {
    off = (off + 255) & ~(size_t)255;
    void* p = (char*)d_ws + off;
    off += bytes;
    return p;
  };
  f16*   xn_h    = (f16*)  alloc(MR * D * sizeof(f16));
  f16*   fc1w_h  = (f16*)  alloc((size_t)ED * D * sizeof(f16));
  f16*   h_h     = (f16*)  alloc(MR * ED * sizeof(f16));
  f16*   c1t_h   = (f16*)  alloc((size_t)FEAT * ED * sizeof(f16));   // [512,4096]
  float* act     = (float*)alloc(MR * 512 * sizeof(float));
  float* att     = (float*)alloc(MR * 8 * sizeof(float));
  f16*   actT_h  = (f16*)  alloc((size_t)Bb * 64 * 2400 * sizeof(f16));
  float* asum    = (float*)alloc((size_t)Bb * 64 * sizeof(float));
  f16*   resT_h  = (f16*)  alloc((size_t)Bb * 512 * 2400 * sizeof(f16));
  float* vlad    = (float*)alloc((size_t)Bb * 512 * 64 * sizeof(float));
  f16*   vlad_h  = (f16*)  alloc((size_t)Bb * VLAD * sizeof(f16));
  f16*   cgfw_h  = (f16*)  alloc((size_t)H * VLAD * sizeof(f16));
  float* y_f     = (float*)alloc((size_t)Bb * H * sizeof(float));
  f16*   y_h     = (f16*)  alloc((size_t)Bb * H * sizeof(f16));
  f16*   g1w_h   = (f16*)  alloc((size_t)HR * H * sizeof(f16));
  f16*   g2w_h   = (f16*)  alloc((size_t)H * HR * sizeof(f16));
  f16*   z_h     = (f16*)  alloc((size_t)Bb * HR * sizeof(f16));
  float* gate    = (float*)alloc((size_t)Bb * H * sizeof(float));
  f16*   yg_h    = (f16*)  alloc((size_t)Bb * H * sizeof(f16));
  f16*   fc3w_h  = (f16*)  alloc((size_t)NC * H * sizeof(f16));
  (void)ws_size; (void)in_sizes; (void)n_in; (void)out_size;

  const dim3 blk(256);

  // 1) per-frame L2 normalize -> f16
  l2norm_rows<<<(unsigned)MR, blk, 0, stream>>>(x, xn_h, D);

  // 2) weight conversions
  cvt_f16_kernel<<<2048, blk, 0, stream>>>(fc1_w, fc1w_h, (long)ED * D);
  transpose_cvt_kernel<<<2048, blk, 0, stream>>>(c1, c1t_h, ED, FEAT); // c1[4096,512] -> [512,4096]
  cvt_f16_kernel<<<2048, blk, 0, stream>>>(cgf_w, cgfw_h, (long)H * VLAD);
  cvt_f16_kernel<<<512, blk, 0, stream>>>(g1_w, g1w_h, (long)HR * H);
  cvt_f16_kernel<<<512, blk, 0, stream>>>(g2_w, g2w_h, (long)H * HR);
  cvt_f16_kernel<<<2048, blk, 0, stream>>>(fc3_w, fc3w_h, (long)NC * H);

  // 3) h = xn @ fc1_w^T + fc1_b   [9600,4096] f16
  gemm_nt_wmma<<<dim3(75, 32, 1), blk, 0, stream>>>(
      xn_h, fc1w_h, fc1_b, nullptr, nullptr, nullptr, nullptr,
      nullptr, nullptr, nullptr, h_h, (int)MR, ED, D, 0, 0, 0, 0);

  // 4) attention  [9600,8]
  att_kernel<<<(unsigned)MR, blk, 0, stream>>>(h_h, fc2_w, fc2_b, att, ED);

  // 5) act = bn1(h @ c1)   [9600,512] f32
  gemm_nt_wmma<<<dim3(75, 4, 1), blk, 0, stream>>>(
      h_h, c1t_h, nullptr, bn1_g, bn1_b, bn1_m, bn1_v,
      nullptr, nullptr, act, nullptr, (int)MR, FEAT, ED, 0, 0, 0, 0);

  // 6) softmax over K * attention (in-place) + build actT
  softmax_att_kernel<<<(unsigned)MR, blk, 0, stream>>>(act, att, actT_h);

  // 7) asum[b,k]
  colsum_kernel<<<256, blk, 0, stream>>>(act, asum);

  // 8) resT[b][f][n]
  build_resT_kernel<<<8192, blk, 0, stream>>>(h_h, resT_h);

  // 9) vlad[b,f,k] = resT[b] @ actT[b]^T - asum*c2   (batched WMMA GEMM)
  gemm_nt_wmma<<<dim3(4, 1, 32), blk, 0, stream>>>(
      resT_h, actT_h, nullptr, nullptr, nullptr, nullptr, nullptr,
      asum, c2, vlad, nullptr, 512, 64, 2400,
      (long)512 * 2400, (long)64 * 2400, (long)512 * 64, 0);

  // 10) intra-norm + bn2 -> vlad_h [32,32768] f16
  intranorm_kernel<<<256, blk, 0, stream>>>(vlad, bn2_g, bn2_b, bn2_m, bn2_v, vlad_h);

  // 11) y = cgbn(vlad @ cgf_w^T + cgf_b)  [32,2048]
  gemm_nt_wmma<<<dim3(1, 16, 1), blk, 0, stream>>>(
      vlad_h, cgfw_h, cgf_b, cgbn_g, cgbn_b, cgbn_m, cgbn_v,
      nullptr, nullptr, y_f, y_h, Bb, H, VLAD, 0, 0, 0, 0);

  // 12) z = relu(gbn(y @ g1_w^T + g1_b))  [32,256] f16
  gemm_nt_wmma<<<dim3(1, 2, 1), blk, 0, stream>>>(
      y_h, g1w_h, g1_b, gbn_g, gbn_b, gbn_m, gbn_v,
      nullptr, nullptr, nullptr, z_h, Bb, HR, H, 0, 0, 0, FLAG_RELU);

  // 13) gate = sigmoid(z @ g2_w^T + g2_b)  [32,2048] f32
  gemm_nt_wmma<<<dim3(1, 16, 1), blk, 0, stream>>>(
      z_h, g2w_h, g2_b, nullptr, nullptr, nullptr, nullptr,
      nullptr, nullptr, gate, nullptr, Bb, H, HR, 0, 0, 0, FLAG_SIG);

  // 14) yg = y * gate -> f16
  mulcvt_kernel<<<256, blk, 0, stream>>>(y_f, gate, yg_h, (long)Bb * H);

  // 15) out = yg @ fc3_w^T + fc3_b  [32,3862] f32
  gemm_nt_wmma<<<dim3(1, 31, 1), blk, 0, stream>>>(
      yg_h, fc3w_h, fc3_b, nullptr, nullptr, nullptr, nullptr,
      nullptr, nullptr, (float*)d_out, nullptr, Bb, NC, H, 0, 0, 0, 0);
}
